// DiTBlock_61684320305745
// MI455X (gfx1250) — matrix operand order repaired and checked by reading, not compiled, not run on hardware
//
#include <hip/hip_runtime.h>
#include <math.h>

// ---------------------------------------------------------------------------
// Types for CDNA5 WMMA (wave32): v_wmma_f32_16x16x32_bf16
// ---------------------------------------------------------------------------
typedef __attribute__((ext_vector_type(16))) __bf16 v16bf;
typedef __attribute__((ext_vector_type(8)))  float  v8f;
typedef __attribute__((ext_vector_type(4)))  unsigned int u32x4;
typedef int async_i32x4 __attribute__((vector_size(16)));  // matches builtin param

#define D_MODEL 512
#define T_SEQ   2048
#define NHEAD   8
#define DK      64
#define DFF     2048

// ---- CDNA5 async global->LDS copy (ASYNCcnt path), guarded by __has_builtin
#if defined(__HIP_DEVICE_COMPILE__) && defined(__has_builtin)
#if __has_builtin(__builtin_amdgcn_global_load_async_to_lds_b128)
#define HAVE_ASYNC_LDS 1
#endif
#endif

#ifdef HAVE_ASYNC_LDS
// signature (from clang diagnostic): (int4 AS1* src, int4 AS3* dst, Ii off, Ii cpol)
#define ASYNC_CP16(dst_lds, src_glob)                                          \
    __builtin_amdgcn_global_load_async_to_lds_b128(                            \
        (__attribute__((address_space(1))) async_i32x4*)(void*)(src_glob),     \
        (__attribute__((address_space(3))) async_i32x4*)(void*)(dst_lds), 0, 0)
#if __has_builtin(__builtin_amdgcn_s_wait_asynccnt)
#define WAIT_ASYNC(n) __builtin_amdgcn_s_wait_asynccnt(n)
#else
#define WAIT_ASYNC(n) asm volatile("s_wait_asynccnt %0" ::"i"(n) : "memory")
#endif
#endif

__device__ __forceinline__ unsigned short f2bf(float f) {
    unsigned int u = __float_as_uint(f);
    unsigned int r = u + 0x7FFFu + ((u >> 16) & 1u);
    return (unsigned short)(r >> 16);
}

// A fragment: 16x32 bf16 tile, row-major source with row stride ld (elements).
// Lane L holds row L%16; K chunks at 8*(L/16) and 16+8*(L/16)  (ISA 7.12.2).
__device__ __forceinline__ v16bf load_frag_a(const unsigned short* base, int ld) {
    int lane = threadIdx.x & 31;
    const unsigned short* p = base + (lane & 15) * ld + ((lane >> 4) << 3);
    union { u32x4 q[2]; v16bf v; } u;
    u.q[0] = *(const u32x4*)(p);
    u.q[1] = *(const u32x4*)(p + 16);
    return u.v;
}

// B fragment: 32x16 bf16 tile (KxN), row-major source with row stride ld.
// Lane L holds column N=L%16; K at 8*(L/16)+i and 16+8*(L/16)+i (matches A).
__device__ __forceinline__ v16bf load_frag_b(const unsigned short* base, int ld) {
    int lane = threadIdx.x & 31;
    const unsigned short* p = base + (lane & 15) + ((lane >> 4) << 3) * ld;
    union { unsigned short s[16]; v16bf v; } u;
#pragma unroll
    for (int i = 0; i < 8; ++i) {
        u.s[i]     = p[i * ld];
        u.s[8 + i] = p[(16 + i) * ld];
    }
    return u.v;
}

__device__ __forceinline__ v8f wmma_bf16(v16bf a, v16bf b, v8f c) {
    return __builtin_amdgcn_wmma_f32_16x16x32_bf16(false, a, false, b, (short)0, c,
                                                   false, false);
}

// ---------------------------------------------------------------------------
// f32 -> bf16 weight conversion
// ---------------------------------------------------------------------------
__global__ void cvt_bf16_kernel(const float* __restrict__ src,
                                unsigned short* __restrict__ dst, int n) {
    for (int i = blockIdx.x * blockDim.x + threadIdx.x; i < n;
         i += gridDim.x * blockDim.x)
        dst[i] = f2bf(src[i]);
}

// ---------------------------------------------------------------------------
// mod[b, 0:4D] = silu(c[b]) @ W_ada + b_ada      (tiny: 4 x 2048 outputs)
// ---------------------------------------------------------------------------
__global__ __launch_bounds__(256) void silu_ada_kernel(
    const float* __restrict__ c, const float* __restrict__ Wada,
    const float* __restrict__ bada, float* __restrict__ mod) {
    __shared__ float sc[D_MODEL];
    int b = blockIdx.x, tid = threadIdx.x;
    float c0 = c[b * D_MODEL + tid];
    float c1 = c[b * D_MODEL + 256 + tid];
    sc[tid]       = c0 / (1.0f + __expf(-c0));
    sc[tid + 256] = c1 / (1.0f + __expf(-c1));
    __syncthreads();
    int j = blockIdx.y * 256 + tid;
    float acc = bada[j];
    for (int d = 0; d < D_MODEL; ++d)
        acc = fmaf(sc[d], Wada[d * (4 * D_MODEL) + j], acc);
    mod[b * (4 * D_MODEL) + j] = acc;
}

// ---------------------------------------------------------------------------
// LayerNorm + adaLN scale -> bf16:  out = (LN(x)*g+b) * (1 + mod[b, moff+d])
// ---------------------------------------------------------------------------
__global__ __launch_bounds__(256) void ln_mod_kernel(
    const float* __restrict__ x, const float* __restrict__ g,
    const float* __restrict__ be, const float* __restrict__ mod, int moff,
    unsigned short* __restrict__ out) {
    __shared__ float sbuf[8], qbuf[8];
    int row = blockIdx.x, tid = threadIdx.x;
    int wave = tid >> 5, lane = tid & 31;
    const float* xr = x + row * D_MODEL;
    float a0 = xr[tid], a1 = xr[tid + 256];
    float s = a0 + a1, q = a0 * a0 + a1 * a1;
#pragma unroll
    for (int m = 16; m; m >>= 1) {
        s += __shfl_xor(s, m, 32);
        q += __shfl_xor(q, m, 32);
    }
    if (lane == 0) { sbuf[wave] = s; qbuf[wave] = q; }
    __syncthreads();
    float ts = 0.f, tq = 0.f;
#pragma unroll
    for (int i = 0; i < 8; ++i) { ts += sbuf[i]; tq += qbuf[i]; }
    float mu = ts * (1.0f / D_MODEL);
    float var = tq * (1.0f / D_MODEL) - mu * mu;
    float rs = rsqrtf(var + 1e-5f);
    int b = row >> 11;
    const float* mrow = mod + b * (4 * D_MODEL) + moff;
    int d0 = tid, d1 = tid + 256;
    float y0 = ((a0 - mu) * rs * g[d0] + be[d0]) * (1.0f + mrow[d0]);
    float y1 = ((a1 - mu) * rs * g[d1] + be[d1]) * (1.0f + mrow[d1]);
    out[row * D_MODEL + d0] = f2bf(y0);
    out[row * D_MODEL + d1] = f2bf(y1);
}

// ---------------------------------------------------------------------------
// Generic bf16 WMMA GEMM:  C[M,N] = A[M,K] @ W[K,N]   (A,W bf16, acc f32)
// 256 threads = 8 waves; block tile 256(M) x 64(N); wave tile 32 x 64.
// B tile (32x64) triple-buffered through LDS via GLOBAL_LOAD_ASYNC_TO_LDS
// (ASYNCcnt) when available; synchronous staging fallback otherwise.
// Epilogues:
//   1: Q proj   -> bf16 [B,H,T,dk], scaled by 1/sqrt(dk)
//   2: K proj   -> bf16 [B,H,dk,T]  (transposed for score B-fragments)
//   3: V proj   -> bf16 [B,H,T,dk]
//   4: O proj   -> f32  x1 = resid + gate[b,n]*acc
//   5: MLP1     -> bf16 gelu(acc + bias[n])     (exact erf gelu)
//   6: MLP2     -> f32  out = resid + gate[b,n]*(acc + bias[n])
// ---------------------------------------------------------------------------
__global__ __launch_bounds__(256) void gemm_bf16_kernel(
    const unsigned short* __restrict__ A, const unsigned short* __restrict__ W,
    int M, int N, int K, int epi,
    const float* __restrict__ bias, const float* __restrict__ gate,
    const float* __restrict__ resid, float* __restrict__ outf,
    unsigned short* __restrict__ outh) {
#ifdef HAVE_ASYNC_LDS
    __shared__ __align__(16) unsigned short bs[3][32 * 64];  // 12 KB, 3 buffers
#else
    __shared__ __align__(16) unsigned short bs[1][32 * 64];  // 4 KB
#endif
    int tid = threadIdx.x;
    int wave = tid >> 5, lane = tid & 31;
    int n0 = blockIdx.x * 64;
    int m0 = blockIdx.y * 256 + wave * 32;

    v8f acc[2][4];
    const v8f vzero = {0.f, 0.f, 0.f, 0.f, 0.f, 0.f, 0.f, 0.f};
#pragma unroll
    for (int i = 0; i < 2; ++i)
#pragma unroll
        for (int j = 0; j < 4; ++j) acc[i][j] = vzero;

    // this thread's 16B slice of the 32x64 B tile
    int e = tid * 8;
    int br = e >> 6, bc = e & 63;
    const unsigned short* wsrc = W + br * N + n0 + bc;  // + k0*N per step
    int nk = K >> 5;

#ifdef HAVE_ASYNC_LDS
    ASYNC_CP16(&bs[0][e], wsrc);  // preload tile 0
    for (int i = 0; i < nk; ++i) {
        if (i + 1 < nk) {
            ASYNC_CP16(&bs[(i + 1) % 3][e], wsrc + (size_t)(i + 1) * 32 * N);
            WAIT_ASYNC(1);  // in-order completion: tile i done
        } else {
            WAIT_ASYNC(0);
        }
        __syncthreads();
        const unsigned short* bt = &bs[i % 3][0];
        int k0 = i << 5;
#pragma unroll
        for (int mt = 0; mt < 2; ++mt) {
            v16bf af = load_frag_a(A + (m0 + mt * 16) * K + k0, K);
#pragma unroll
            for (int nt = 0; nt < 4; ++nt) {
                v16bf bf = load_frag_b(bt + nt * 16, 64);
                acc[mt][nt] = wmma_bf16(af, bf, acc[mt][nt]);
            }
        }
    }
#else
    for (int i = 0; i < nk; ++i) {
        int k0 = i << 5;
        *(u32x4*)&bs[0][e] = *(const u32x4*)(wsrc + (size_t)k0 * N);
        __syncthreads();
#pragma unroll
        for (int mt = 0; mt < 2; ++mt) {
            v16bf af = load_frag_a(A + (m0 + mt * 16) * K + k0, K);
#pragma unroll
            for (int nt = 0; nt < 4; ++nt) {
                v16bf bf = load_frag_b(&bs[0][nt * 16], 64);
                acc[mt][nt] = wmma_bf16(af, bf, acc[mt][nt]);
            }
        }
        __syncthreads();
    }
#endif

    int half = lane >> 4;
    int nl = lane & 15;
#pragma unroll
    for (int mt = 0; mt < 2; ++mt)
#pragma unroll
        for (int nt = 0; nt < 4; ++nt)
#pragma unroll
            for (int r = 0; r < 8; ++r) {
                int m = m0 + mt * 16 + r + 8 * half;
                int n = n0 + nt * 16 + nl;
                float aV = acc[mt][nt][r];
                if (epi == 1) {  // Q: [B,H,T,dk] with 1/sqrt(dk)
                    int b = m >> 11, t = m & 2047, h = n >> 6, d = n & 63;
                    outh[(((b * NHEAD + h) * T_SEQ) + t) * DK + d] =
                        f2bf(aV * 0.125f);
                } else if (epi == 2) {  // K transposed: [B,H,dk,T]
                    int b = m >> 11, t = m & 2047, h = n >> 6, d = n & 63;
                    outh[((b * NHEAD + h) * DK + d) * T_SEQ + t] = f2bf(aV);
                } else if (epi == 3) {  // V: [B,H,T,dk]
                    int b = m >> 11, t = m & 2047, h = n >> 6, d = n & 63;
                    outh[(((b * NHEAD + h) * T_SEQ) + t) * DK + d] = f2bf(aV);
                } else if (epi == 4) {  // O proj: x1 = x + gate_msa * acc
                    int b = m >> 11;
                    outf[m * D_MODEL + n] =
                        resid[m * D_MODEL + n] + gate[b * (4 * D_MODEL) + n] * aV;
                } else if (epi == 5) {  // MLP1: gelu(acc + bias) -> bf16
                    float hv = aV + bias[n];
                    float ge = 0.5f * hv * (1.0f + erff(hv * 0.70710678118654752f));
                    outh[m * N + n] = f2bf(ge);
                } else {  // epi == 6: out = x1 + gate_mlp * (acc + bias)
                    int b = m >> 11;
                    outf[m * D_MODEL + n] =
                        resid[m * D_MODEL + n] +
                        gate[b * (4 * D_MODEL) + n] * (aV + bias[n]);
                }
            }
}

// ---------------------------------------------------------------------------
// Flash attention, one wave per (b, h, 16-query tile).
// q: [B,H,T,dk] bf16 (pre-scaled), kT: [B,H,dk,T] bf16, v: [B,H,T,dk] bf16
// out: [B,T, H*dk] bf16 (token-major for the O projection GEMM)
// ---------------------------------------------------------------------------
__global__ __launch_bounds__(32) void attn_kernel(
    const unsigned short* __restrict__ q, const unsigned short* __restrict__ kT,
    const unsigned short* __restrict__ v, unsigned short* __restrict__ out) {
    __shared__ __align__(16) unsigned short plds[16 * 32];  // P tile bf16
    int lane = threadIdx.x;
    int qt = blockIdx.x, h = blockIdx.y, b = blockIdx.z;
    int bh = b * NHEAD + h;

    const unsigned short* qbase = q + (bh * T_SEQ + qt * 16) * DK;
    v16bf aq0 = load_frag_a(qbase, DK);        // dk 0..31
    v16bf aq1 = load_frag_a(qbase + 32, DK);   // dk 32..63

    const unsigned short* ktb = kT + bh * DK * T_SEQ;
    const unsigned short* vb  = v + bh * T_SEQ * DK;

    const v8f vzero = {0.f, 0.f, 0.f, 0.f, 0.f, 0.f, 0.f, 0.f};
    v8f o[4];
#pragma unroll
    for (int i = 0; i < 4; ++i) o[i] = vzero;
    float rmax[8], rsum[8];
#pragma unroll
    for (int r = 0; r < 8; ++r) { rmax[r] = -1e30f; rsum[r] = 0.f; }

    int half = lane >> 4, nl = lane & 15;

    for (int kb = 0; kb < T_SEQ; kb += 32) {
        // S tile 16 x 32 : two 16x16 fragments, K-dim = dk = 64 (2 WMMA each)
        v8f s0 = vzero, s1 = vzero;
        v16bf bk;
        bk = load_frag_b(ktb + kb, T_SEQ);                  s0 = wmma_bf16(aq0, bk, s0);
        bk = load_frag_b(ktb + 32 * T_SEQ + kb, T_SEQ);     s0 = wmma_bf16(aq1, bk, s0);
        bk = load_frag_b(ktb + kb + 16, T_SEQ);             s1 = wmma_bf16(aq0, bk, s1);
        bk = load_frag_b(ktb + 32 * T_SEQ + kb + 16, T_SEQ); s1 = wmma_bf16(aq1, bk, s1);

        // online softmax over this 32-key block (row stats across 16 lanes)
#pragma unroll
        for (int r = 0; r < 8; ++r) {
            float mx = fmaxf(s0[r], s1[r]);
#pragma unroll
            for (int msk = 8; msk; msk >>= 1) mx = fmaxf(mx, __shfl_xor(mx, msk, 32));
            float nm = fmaxf(rmax[r], mx);
            float al = __expf(rmax[r] - nm);
            float p0 = __expf(s0[r] - nm);
            float p1 = __expf(s1[r] - nm);
            float ps = p0 + p1;
#pragma unroll
            for (int msk = 8; msk; msk >>= 1) ps += __shfl_xor(ps, msk, 32);
            rsum[r] = rsum[r] * al + ps;
            rmax[r] = nm;
            o[0][r] *= al; o[1][r] *= al; o[2][r] *= al; o[3][r] *= al;
            int m = r + 8 * half;
            plds[m * 32 + nl]      = f2bf(p0);
            plds[m * 32 + 16 + nl] = f2bf(p1);
        }
        asm volatile("s_wait_dscnt 0" ::: "memory");

        // O += P(16x32) @ V(32x64)
        v16bf ap = load_frag_a(plds, 32);
        v16bf bv;
        bv = load_frag_b(vb + kb * DK + 0,  DK); o[0] = wmma_bf16(ap, bv, o[0]);
        bv = load_frag_b(vb + kb * DK + 16, DK); o[1] = wmma_bf16(ap, bv, o[1]);
        bv = load_frag_b(vb + kb * DK + 32, DK); o[2] = wmma_bf16(ap, bv, o[2]);
        bv = load_frag_b(vb + kb * DK + 48, DK); o[3] = wmma_bf16(ap, bv, o[3]);
    }

    float inv[8];
#pragma unroll
    for (int r = 0; r < 8; ++r) inv[r] = 1.0f / rsum[r];
#pragma unroll
    for (int nt = 0; nt < 4; ++nt)
#pragma unroll
        for (int r = 0; r < 8; ++r) {
            int m = r + 8 * half;
            int t = qt * 16 + m;
            out[(b * T_SEQ + t) * D_MODEL + h * DK + nt * 16 + nl] =
                f2bf(o[nt][r] * inv[r]);
        }
}

// ---------------------------------------------------------------------------
// Host launcher
// ---------------------------------------------------------------------------
extern "C" void kernel_launch(void* const* d_in, const int* in_sizes, int n_in,
                              void* d_out, int out_size, void* d_ws, size_t ws_size,
                              hipStream_t stream) {
    const float* x     = (const float*)d_in[0];
    const float* c     = (const float*)d_in[1];
    const float* Wq    = (const float*)d_in[2];
    const float* Wk    = (const float*)d_in[3];
    const float* Wv    = (const float*)d_in[4];
    const float* Wo    = (const float*)d_in[5];
    const float* g1    = (const float*)d_in[6];
    const float* b1    = (const float*)d_in[7];
    const float* g2    = (const float*)d_in[8];
    const float* b2    = (const float*)d_in[9];
    const float* Wm1   = (const float*)d_in[10];
    const float* bm1   = (const float*)d_in[11];
    const float* Wm2   = (const float*)d_in[12];
    const float* bm2   = (const float*)d_in[13];
    const float* Wada  = (const float*)d_in[14];
    const float* bada  = (const float*)d_in[15];
    float* outp = (float*)d_out;

    const int B = 4, M = B * T_SEQ;  // 8192 tokens
    char* ws = (char*)d_ws;
    size_t off = 0;
    auto alloc = [&](size_t bytes) { char* p = ws + off; off = (off + bytes + 255) & ~(size_t)255; return p; };

    float*          modp   = (float*)         alloc((size_t)B * 4 * D_MODEL * 4);
    unsigned short* Wq_b   = (unsigned short*)alloc((size_t)D_MODEL * D_MODEL * 2);
    unsigned short* Wk_b   = (unsigned short*)alloc((size_t)D_MODEL * D_MODEL * 2);
    unsigned short* Wv_b   = (unsigned short*)alloc((size_t)D_MODEL * D_MODEL * 2);
    unsigned short* Wo_b   = (unsigned short*)alloc((size_t)D_MODEL * D_MODEL * 2);
    unsigned short* Wm1_b  = (unsigned short*)alloc((size_t)D_MODEL * DFF * 2);
    unsigned short* Wm2_b  = (unsigned short*)alloc((size_t)DFF * D_MODEL * 2);
    unsigned short* xn     = (unsigned short*)alloc((size_t)M * D_MODEL * 2);
    unsigned short* qb     = (unsigned short*)alloc((size_t)M * D_MODEL * 2);
    unsigned short* kTb    = (unsigned short*)alloc((size_t)M * D_MODEL * 2);
    unsigned short* vbuf   = (unsigned short*)alloc((size_t)M * D_MODEL * 2);
    unsigned short* attnb  = (unsigned short*)alloc((size_t)M * D_MODEL * 2);
    float*          x1     = (float*)         alloc((size_t)M * D_MODEL * 4);
    unsigned short* hmid   = (unsigned short*)alloc((size_t)M * DFF * 2);

    // weights -> bf16
    cvt_bf16_kernel<<<256, 256, 0, stream>>>(Wq,  Wq_b,  D_MODEL * D_MODEL);
    cvt_bf16_kernel<<<256, 256, 0, stream>>>(Wk,  Wk_b,  D_MODEL * D_MODEL);
    cvt_bf16_kernel<<<256, 256, 0, stream>>>(Wv,  Wv_b,  D_MODEL * D_MODEL);
    cvt_bf16_kernel<<<256, 256, 0, stream>>>(Wo,  Wo_b,  D_MODEL * D_MODEL);
    cvt_bf16_kernel<<<512, 256, 0, stream>>>(Wm1, Wm1_b, D_MODEL * DFF);
    cvt_bf16_kernel<<<512, 256, 0, stream>>>(Wm2, Wm2_b, DFF * D_MODEL);

    // adaLN modulation
    silu_ada_kernel<<<dim3(B, 8), 256, 0, stream>>>(c, Wada, bada, modp);

    // xn1 = LN(x) * (1 + scale_msa)
    ln_mod_kernel<<<M, 256, 0, stream>>>(x, g1, b1, modp, 0, xn);

    // QKV projections  (block tile 256x64)
    dim3 gq(D_MODEL / 64, M / 256);
    gemm_bf16_kernel<<<gq, 256, 0, stream>>>(xn, Wq_b, M, D_MODEL, D_MODEL, 1,
                                             nullptr, nullptr, nullptr, nullptr, qb);
    gemm_bf16_kernel<<<gq, 256, 0, stream>>>(xn, Wk_b, M, D_MODEL, D_MODEL, 2,
                                             nullptr, nullptr, nullptr, nullptr, kTb);
    gemm_bf16_kernel<<<gq, 256, 0, stream>>>(xn, Wv_b, M, D_MODEL, D_MODEL, 3,
                                             nullptr, nullptr, nullptr, nullptr, vbuf);

    // attention
    attn_kernel<<<dim3(T_SEQ / 16, NHEAD, B), 32, 0, stream>>>(qb, kTb, vbuf, attnb);

    // x1 = x + gate_msa * (attn @ Wo)
    gemm_bf16_kernel<<<gq, 256, 0, stream>>>(attnb, Wo_b, M, D_MODEL, D_MODEL, 4,
                                             nullptr, modp + D_MODEL, x, x1, nullptr);

    // xn2 = LN(x1) * (1 + scale_mlp)
    ln_mod_kernel<<<M, 256, 0, stream>>>(x1, g2, b2, modp, 2 * D_MODEL, xn);

    // h = gelu(xn2 @ W_mlp1 + b_mlp1)
    dim3 g1d(DFF / 64, M / 256);
    gemm_bf16_kernel<<<g1d, 256, 0, stream>>>(xn, Wm1_b, M, DFF, D_MODEL, 5,
                                              bm1, nullptr, nullptr, nullptr, hmid);

    // out = x1 + gate_mlp * (h @ W_mlp2 + b_mlp2)
    gemm_bf16_kernel<<<gq, 256, 0, stream>>>(hmid, Wm2_b, M, D_MODEL, DFF, 6,
                                             bm2, modp + 3 * D_MODEL, x1, outp, nullptr);
}
